// Node_Readout_atom_embedding_only_87634512707841
// MI455X (gfx1250) — matrix operand order repaired and checked by reading, not compiled, not run on hardware
//
#include <hip/hip_runtime.h>
#include <hip/hip_bf16.h>
#include <stdint.h>

// ---------------------------------------------------------------------------
// Problem constants (from reference)
// ---------------------------------------------------------------------------
#define HID 512
#define D_FF 2048
#define ATOM_FDIM 151
#define MAX_NBR 8
#define N_MOLS 1024
#define ATOMS_PER_MOL 24
#define N_ATOMS (1 + N_MOLS * ATOMS_PER_MOL)   // 24577
#define FEAT_DIM 200
#define FFN_HIDDEN 1024
#define NUM_TASKS 12

#define M_PAD 24592          // N_ATOMS padded up to multiple of 16
#define KX 672               // 151+512 = 663 padded to multiple of 32
#define KMOL 736             // 512+200 = 712 padded to multiple of 32
#define LN_EPS 1e-6f

typedef __attribute__((ext_vector_type(16))) __bf16 v16bf;
typedef __attribute__((ext_vector_type(8)))  float  v8f;

__device__ __forceinline__ unsigned short f2bf(float x) {
  union { float f; unsigned u; } v; v.f = x;
  unsigned r = v.u + 0x7FFFu + ((v.u >> 16) & 1u);   // round-to-nearest-even
  return (unsigned short)(r >> 16);
}

// ---------------------------------------------------------------------------
// Stage 1: neighbor gather-sum + concat(original_f_atoms, aggr) -> bf16 A mat
//   X row layout: [0,151) = f_atoms, [151,663) = sum_j feat[nbr[j]], [663,672)=0
//   128 threads / row; aggregation done as float4 gathers (global_load_b128).
// ---------------------------------------------------------------------------
__global__ void build_x_kernel(const float* __restrict__ fatoms,
                               const float* __restrict__ feat,
                               const int*   __restrict__ nbr,
                               unsigned short* __restrict__ X,
                               int rows_valid)
{
  const int row = blockIdx.x;
  const int tid = threadIdx.x;      // 128
  unsigned short* xr = X + (size_t)row * KX;
  if (row >= rows_valid) {
    for (int c = tid; c < KX; c += 128) xr[c] = 0;
    return;
  }
  // features [0,151)
  xr[tid] = f2bf(fatoms[(size_t)row * ATOM_FDIM + tid]);
  if (tid + 128 < ATOM_FDIM)
    xr[tid + 128] = f2bf(fatoms[(size_t)row * ATOM_FDIM + tid + 128]);
  // tail zero pad [663,672)
  if (tid < KX - (ATOM_FDIM + HID)) xr[ATOM_FDIM + HID + tid] = 0;

  // neighbor gather-sum: 4 contiguous H-columns per thread
  int idx[MAX_NBR];
#pragma unroll
  for (int j = 0; j < MAX_NBR; ++j) idx[j] = nbr[(size_t)row * MAX_NBR + j];

  const int hcol = tid * 4;
  float sx = 0.f, sy = 0.f, sz = 0.f, sw = 0.f;
#pragma unroll
  for (int j = 0; j < MAX_NBR; ++j) {
    const float4 v = *(const float4*)(feat + (size_t)idx[j] * HID + hcol);
    sx += v.x; sy += v.y; sz += v.z; sw += v.w;
  }
  unsigned short* agg = xr + ATOM_FDIM + hcol;
  agg[0] = f2bf(sx); agg[1] = f2bf(sy); agg[2] = f2bf(sz); agg[3] = f2bf(sw);
}

// ---------------------------------------------------------------------------
// Weight f32[din,dout] -> TRANSPOSED bf16 [npad, kpad] (row n = output col),
// zero padded. Transposing here makes GEMM B staging identical to A staging.
// ---------------------------------------------------------------------------
__global__ void convert_weight_t_kernel(const float* __restrict__ src,
                                        unsigned short* __restrict__ dst,
                                        int din, int dout, int kpad, int npad)
{
  const int t = blockIdx.x * blockDim.x + threadIdx.x;
  if (t >= npad * kpad) return;
  const int n = t / kpad, k = t % kpad;     // consecutive t -> consecutive k (coalesced store)
  const float v = (k < din && n < dout) ? src[(size_t)k * dout + n] : 0.f;
  dst[t] = f2bf(v);
}

// ---------------------------------------------------------------------------
// Tiled bf16 WMMA GEMM: C[M,N] = act(A[M,K] @ Bt[N,K]^T + bias)
//   Block tile 128x128, BK=32, double-buffered LDS (one barrier per K step).
//   256 threads = 8 waves (2x4), each wave owns a 64x32 micro tile =
//   4x2 v8f accumulators -> 8 v_wmma_f32_16x16x32_bf16 per step.
//   All 6 fragments are hoisted before the WMMA chain so the 12 ds_load_b128
//   issue together and the matrix pipe runs without LDS-latency stalls.
// ---------------------------------------------------------------------------
template<int RELU, int OUT_BF16>
__launch_bounds__(256)
__global__ void gemm_wmma_bf16(const unsigned short* __restrict__ A,   // [M, lda]
                               const unsigned short* __restrict__ Bt,  // [Nb, ldbt]
                               const float* __restrict__ bias,
                               void* __restrict__ Cout,
                               int M, int N, int K,
                               int lda, int ldbt, int Nb, int ldc)
{
  __shared__ unsigned short As[2][128][40];   // [buf][m][k], +8 pad
  __shared__ unsigned short Bs[2][128][40];   // [buf][n][k], +8 pad

  const int mBase = blockIdx.y * 128;
  const int nBase = blockIdx.x * 128;
  const int tid   = threadIdx.x;
  const int lane  = tid & 31;
  const int wave  = tid >> 5;
  const int wm    = wave >> 2;   // 0..1  -> 64-row slab
  const int wn    = wave & 3;    // 0..3  -> 32-col slab
  const int lrow  = lane & 15;
  const int khalf = lane >> 4;   // K-half carried by this lane (contiguous k = 16*khalf + s)

  // staging coords: each thread moves 16 bf16 of A and 16 bf16 of B per step
  const int  sr   = tid >> 1;                 // 0..127 (tile row)
  const int  sh   = tid & 1;                  // 16-elem half of BK
  const int  aRow = mBase + sr;
  const int  bRow = nBase + sr;
  const bool aOk  = (aRow < M);
  const bool bOk  = (bRow < Nb);
  const unsigned short* aSrc = A  + (size_t)aRow * lda  + sh * 16;
  const unsigned short* bSrc = Bt + (size_t)bRow * ldbt + sh * 16;

  uint4 a0, a1, b0, b1;

#define LOAD_TILE(kb)                                                          \
  a0 = make_uint4(0, 0, 0, 0); a1 = a0; b0 = a0; b1 = a0;                      \
  if (aOk) { const uint4* p_ = (const uint4*)(aSrc + (kb)); a0 = p_[0]; a1 = p_[1]; } \
  if (bOk) { const uint4* p_ = (const uint4*)(bSrc + (kb)); b0 = p_[0]; b1 = p_[1]; }

#define STORE_TILE(q)                                                          \
  *(uint4*)&As[q][sr][sh * 16]     = a0; *(uint4*)&As[q][sr][sh * 16 + 8] = a1; \
  *(uint4*)&Bs[q][sr][sh * 16]     = b0; *(uint4*)&Bs[q][sr][sh * 16 + 8] = b1;

  const v8f vzero = {0.f,0.f,0.f,0.f,0.f,0.f,0.f,0.f};
  v8f acc[4][2];
#pragma unroll
  for (int mt = 0; mt < 4; ++mt)
#pragma unroll
    for (int nt = 0; nt < 2; ++nt) acc[mt][nt] = vzero;

  LOAD_TILE(0);
  STORE_TILE(0);
  __syncthreads();

  const int nsteps = K >> 5;
  for (int s = 0; s < nsteps; ++s) {
    const int p = s & 1;
    const bool hasNext = (s + 1 < nsteps);
    if (hasNext) {
      LOAD_TILE((s + 1) << 5);                      // global -> regs (overlaps WMMA)
      if (s + 2 < nsteps) {                         // prefetch 2 tiles ahead
        if (aOk) __builtin_prefetch(aSrc + ((s + 2) << 5), 0, 1);
        if (bOk) __builtin_prefetch(bSrc + ((s + 2) << 5), 0, 1);
      }
    }

    // ---- hoist ALL fragments (2xB, 4xA) so ds_loads issue together ----
    v16bf bfrag[2];
#pragma unroll
    for (int nt = 0; nt < 2; ++nt) {
      union { unsigned u[8]; v16bf v; } ub;
      const unsigned short* bp = &Bs[p][wn * 32 + nt * 16 + lrow][khalf * 16];
#pragma unroll
      for (int j = 0; j < 8; ++j) ub.u[j] = *(const unsigned*)(bp + 2 * j);
      bfrag[nt] = ub.v;
    }
    v16bf afrag[4];
#pragma unroll
    for (int mt = 0; mt < 4; ++mt) {
      union { unsigned u[8]; v16bf v; } ua;
      const unsigned short* ap = &As[p][wm * 64 + mt * 16 + lrow][0];
#pragma unroll
      for (int j = 0; j < 4; ++j) ua.u[j]     = *(const unsigned*)(ap + khalf * 16 + 2 * j);
#pragma unroll
      for (int j = 0; j < 4; ++j) ua.u[4 + j] = *(const unsigned*)(ap + 8 + khalf * 16 + 2 * j);
      afrag[mt] = ua.v;
    }

    // ---- 8 back-to-back WMMAs, no LDS dependencies in between ----
#pragma unroll
    for (int mt = 0; mt < 4; ++mt)
#pragma unroll
      for (int nt = 0; nt < 2; ++nt)
        acc[mt][nt] = __builtin_amdgcn_wmma_f32_16x16x32_bf16(
            /*neg_a=*/false, afrag[mt], /*neg_b=*/false, bfrag[nt],
            /*c_mod=*/(short)0, acc[mt][nt], /*reuse_a=*/false, /*reuse_b=*/false);

    if (hasNext) { STORE_TILE(p ^ 1); }             // regs -> other LDS buffer
    __syncthreads();
  }
#undef LOAD_TILE
#undef STORE_TILE

  // ---- epilogue: bias (+ReLU), store f32 or bf16 ----
  const int rbase = mBase + wm * 64;
  const int cbase = nBase + wn * 32;
#pragma unroll
  for (int nt = 0; nt < 2; ++nt) {
    const int col = cbase + nt * 16 + lrow;
    if (col >= N) continue;
    const float bv = bias[col];
#pragma unroll
    for (int mt = 0; mt < 4; ++mt) {
#pragma unroll
      for (int r = 0; r < 8; ++r) {
        const int row = rbase + mt * 16 + khalf * 8 + r;
        if (row >= M) continue;
        float v = acc[mt][nt][r] + bv;
        if (RELU) v = fmaxf(v, 0.f);
        if (OUT_BF16)
          ((unsigned short*)Cout)[(size_t)row * ldc + col] = f2bf(v);
        else
          ((float*)Cout)[(size_t)row * ldc + col] = v;
      }
    }
  }
}

// ---------------------------------------------------------------------------
// LayerNorm over H=512 columns, one wave32 per row (16 elems/lane)
// ---------------------------------------------------------------------------
__global__ void layernorm_kernel(const float* __restrict__ X,
                                 const float* __restrict__ g,
                                 const float* __restrict__ b,
                                 float* __restrict__ Y, int rows)
{
  const int wave = threadIdx.x >> 5;
  const int lane = threadIdx.x & 31;
  const int row  = blockIdx.x * 8 + wave;
  if (row >= rows) return;
  const float* x = X + (size_t)row * HID;

  float vals[16];
  float s = 0.f;
#pragma unroll
  for (int j = 0; j < 16; ++j) { vals[j] = x[lane + 32 * j]; s += vals[j]; }
#pragma unroll
  for (int m = 16; m >= 1; m >>= 1) s += __shfl_xor(s, m, 32);
  const float mean = s * (1.f / (float)HID);

  float vs = 0.f;
#pragma unroll
  for (int j = 0; j < 16; ++j) { const float d = vals[j] - mean; vs += d * d; }
#pragma unroll
  for (int m = 16; m >= 1; m >>= 1) vs += __shfl_xor(vs, m, 32);
  const float inv = rsqrtf(vs * (1.f / (float)HID) + LN_EPS);

  float* y = Y + (size_t)row * HID;
#pragma unroll
  for (int j = 0; j < 16; ++j) {
    const int c = lane + 32 * j;
    y[c] = (vals[j] - mean) * inv * g[c] + b[c];
  }
}

// ---------------------------------------------------------------------------
// Mol readout: mean over 24 contiguous atoms, concat features -> bf16 [1024,736]
// ---------------------------------------------------------------------------
__global__ void mol_readout_kernel(const float* __restrict__ atomLN,
                                   const float* __restrict__ features,
                                   unsigned short* __restrict__ molin)
{
  const int m = blockIdx.x;
  unsigned short* out = molin + (size_t)m * KMOL;
  for (int c = threadIdx.x; c < KMOL; c += blockDim.x) {
    float v;
    if (c < HID) {
      const float* base = atomLN + (size_t)(1 + m * ATOMS_PER_MOL) * HID + c;
      float s = 0.f;
#pragma unroll
      for (int j = 0; j < ATOMS_PER_MOL; ++j) s += base[(size_t)j * HID];
      v = s * (1.f / (float)ATOMS_PER_MOL);
    } else if (c < HID + FEAT_DIM) {
      v = features[(size_t)m * FEAT_DIM + (c - HID)];
    } else {
      v = 0.f;
    }
    out[c] = f2bf(v);
  }
}

// ---------------------------------------------------------------------------
// Host launcher
// ---------------------------------------------------------------------------
extern "C" void kernel_launch(void* const* d_in, const int* in_sizes, int n_in,
                              void* d_out, int out_size, void* d_ws, size_t ws_size,
                              hipStream_t stream)
{
  (void)in_sizes; (void)n_in; (void)out_size; (void)ws_size;

  const float* atom_output = (const float*)d_in[0];
  const float* bond_output = (const float*)d_in[1];
  const float* f_atoms     = (const float*)d_in[2];
  const int*   a2a         = (const int*)d_in[4];
  const int*   a2b         = (const int*)d_in[5];
  const float* features    = (const float*)d_in[9];
  const float* W1aa = (const float*)d_in[10];
  const float* b1aa = (const float*)d_in[11];
  const float* W2aa = (const float*)d_in[12];
  const float* b2aa = (const float*)d_in[13];
  const float* W1ab = (const float*)d_in[14];
  const float* b1ab = (const float*)d_in[15];
  const float* W2ab = (const float*)d_in[16];
  const float* b2ab = (const float*)d_in[17];
  const float* lnaag = (const float*)d_in[18];
  const float* lnaab = (const float*)d_in[19];
  const float* lnabg = (const float*)d_in[20];
  const float* lnabb = (const float*)d_in[21];
  const float* molW1 = (const float*)d_in[22];
  const float* molb1 = (const float*)d_in[23];
  const float* molW2 = (const float*)d_in[24];
  const float* molb2 = (const float*)d_in[25];

  float* out_mol  = (float*)d_out;                                   // [1024,12]
  float* out_bond = (float*)d_out + (size_t)N_MOLS * NUM_TASKS;      // [24577,512]

  // -------- workspace carve-out (X bf16 and H f32 share region R1) --------
  char* w = (char*)d_ws;
  size_t off = 0;
  auto alloc = [&](size_t bytes) -> void* {
    void* p = w + off;
    off += (bytes + 255) & ~(size_t)255;
    return p;
  };
  void* R1 = alloc((size_t)M_PAD * HID * 4);            // >= M_PAD*KX*2
  unsigned short* Xbf  = (unsigned short*)R1;
  float*          Hf32 = (float*)R1;
  unsigned short* Hbf     = (unsigned short*)alloc((size_t)M_PAD * D_FF * 2);
  float*          atomLN  = (float*)alloc((size_t)N_ATOMS * HID * 4);
  unsigned short* W1bf    = (unsigned short*)alloc((size_t)D_FF * KX * 2);       // [2048][672]
  unsigned short* W2bf    = (unsigned short*)alloc((size_t)HID * D_FF * 2);      // [512][2048]
  unsigned short* molW1bf = (unsigned short*)alloc((size_t)FFN_HIDDEN * KMOL * 2); // [1024][736]
  unsigned short* molW2bf = (unsigned short*)alloc((size_t)16 * FFN_HIDDEN * 2);   // [16][1024]
  unsigned short* molin   = (unsigned short*)alloc((size_t)N_MOLS * KMOL * 2);
  unsigned short* molh    = (unsigned short*)alloc((size_t)N_MOLS * FFN_HIDDEN * 2);

  const dim3 gM1((D_FF + 127) / 128, (M_PAD + 127) / 128);
  const dim3 gM2((HID  + 127) / 128, (M_PAD + 127) / 128);

  // ---------------- atom-from-atom path ----------------
  convert_weight_t_kernel<<<(D_FF * KX + 255) / 256, 256, 0, stream>>>(
      W1aa, W1bf, ATOM_FDIM + HID, D_FF, KX, D_FF);
  convert_weight_t_kernel<<<(HID * D_FF + 255) / 256, 256, 0, stream>>>(
      W2aa, W2bf, D_FF, HID, D_FF, HID);
  build_x_kernel<<<M_PAD, 128, 0, stream>>>(f_atoms, atom_output, a2a, Xbf, N_ATOMS);
  gemm_wmma_bf16<1, 1><<<gM1, 256, 0, stream>>>(
      Xbf, W1bf, b1aa, Hbf, M_PAD, D_FF, KX, KX, KX, D_FF, D_FF);
  gemm_wmma_bf16<0, 0><<<gM2, 256, 0, stream>>>(
      Hbf, W2bf, b2aa, Hf32, M_PAD, HID, D_FF, D_FF, D_FF, HID, HID);
  layernorm_kernel<<<(N_ATOMS + 7) / 8, 256, 0, stream>>>(
      Hf32, lnaag, lnaab, atomLN, N_ATOMS);

  // ---------------- atom-from-bond path (reuses weight + X/H buffers) ----
  convert_weight_t_kernel<<<(D_FF * KX + 255) / 256, 256, 0, stream>>>(
      W1ab, W1bf, ATOM_FDIM + HID, D_FF, KX, D_FF);
  convert_weight_t_kernel<<<(HID * D_FF + 255) / 256, 256, 0, stream>>>(
      W2ab, W2bf, D_FF, HID, D_FF, HID);
  build_x_kernel<<<M_PAD, 128, 0, stream>>>(f_atoms, bond_output, a2b, Xbf, N_ATOMS);
  gemm_wmma_bf16<1, 1><<<gM1, 256, 0, stream>>>(
      Xbf, W1bf, b1ab, Hbf, M_PAD, D_FF, KX, KX, KX, D_FF, D_FF);
  gemm_wmma_bf16<0, 0><<<gM2, 256, 0, stream>>>(
      Hbf, W2bf, b2ab, Hf32, M_PAD, HID, D_FF, D_FF, D_FF, HID, HID);
  layernorm_kernel<<<(N_ATOMS + 7) / 8, 256, 0, stream>>>(
      Hf32, lnabg, lnabb, out_bond, N_ATOMS);

  // ---------------- molecule head ----------------
  convert_weight_t_kernel<<<(FFN_HIDDEN * KMOL + 255) / 256, 256, 0, stream>>>(
      molW1, molW1bf, HID + FEAT_DIM, FFN_HIDDEN, KMOL, FFN_HIDDEN);
  convert_weight_t_kernel<<<(16 * FFN_HIDDEN + 255) / 256, 256, 0, stream>>>(
      molW2, molW2bf, FFN_HIDDEN, NUM_TASKS, FFN_HIDDEN, 16);
  mol_readout_kernel<<<N_MOLS, 256, 0, stream>>>(atomLN, features, molin);

  const dim3 gM3((FFN_HIDDEN + 127) / 128, (N_MOLS + 127) / 128);
  gemm_wmma_bf16<1, 1><<<gM3, 256, 0, stream>>>(
      molin, molW1bf, molb1, molh, N_MOLS, FFN_HIDDEN, KMOL, KMOL, KMOL, FFN_HIDDEN, FFN_HIDDEN);

  const dim3 gM4(1, (N_MOLS + 127) / 128);
  gemm_wmma_bf16<0, 0><<<gM4, 256, 0, stream>>>(
      molh, molW2bf, molb2, out_mol, N_MOLS, NUM_TASKS, FFN_HIDDEN,
      FFN_HIDDEN, FFN_HIDDEN, 16, NUM_TASKS);
}
